// MatchingVanilla_86242943304241
// MI455X (gfx1250) — compile-verified, matrix operand order checked by reading, not run on hardware
//
#include <hip/hip_runtime.h>
#include <hip/hip_bf16.h>
#include <math.h>

typedef __attribute__((ext_vector_type(16))) __bf16 v16bf;
typedef __attribute__((ext_vector_type(8)))  __bf16 v8bf;
typedef __attribute__((ext_vector_type(8)))  float  v8f;

#define MBLK 64
#define NBLK 64
#define LDA 40            // padded bf16 row stride (80B, multiple of 16B)
#define SCALE 0.125f      // 1/sqrt(64)

// Load one 16x32 bf16 WMMA A/B fragment row from LDS.
// Lane half=0 covers K {0..7,16..23}, half=1 covers K {8..15,24..31}.
__device__ __forceinline__ v16bf load_frag(const __bf16* p, int half) {
    union { v16bf v; v8bf h[2]; } u;
    u.h[0] = *(const v8bf*)(p + half * 8);
    u.h[1] = *(const v8bf*)(p + 16 + half * 8);
    return u.v;
}

__global__ __launch_bounds__(256) void k_zero(float* __restrict__ p, int n) {
    int i = blockIdx.x * 256 + threadIdx.x;
    if (i < n) p[i] = 0.0f;
}

// ---- convert inputs fp32 -> bf16 (read once; big GEMMs then re-read half the bytes) ----
__global__ __launch_bounds__(256) void k_prep_x(const float* __restrict__ X, __bf16* __restrict__ Xbf) {
    const size_t i = ((size_t)blockIdx.x * 256 + threadIdx.x) * 8;
    const float* s = X + i;
    v8bf o;
    #pragma unroll
    for (int j = 0; j < 8; ++j) o[j] = (__bf16)s[j];
    *(v8bf*)(Xbf + i) = o;
}

// ---- pack weights into bf16 WMMA B-fragment order: [ntile][kstep][lane][16] ----
__global__ __launch_bounds__(256) void k_prep_w(
    const float* __restrict__ Wvc, const float* __restrict__ Wvp,
    const float* __restrict__ Wkc, const float* __restrict__ Wkp,
    const float* __restrict__ Wout,
    __bf16* __restrict__ Pvc, __bf16* __restrict__ Pvp,
    __bf16* __restrict__ Pkc, __bf16* __restrict__ Pkp,
    __bf16* __restrict__ Pout)
{
    const int z = blockIdx.z;
    const float* W; __bf16* P; int KS;
    if      (z == 0) { W = Wvc;  P = Pvc;  KS = 8;  }
    else if (z == 1) { W = Wvp;  P = Pvp;  KS = 8;  }
    else if (z == 2) { W = Wkc;  P = Pkc;  KS = 8;  }
    else if (z == 3) { W = Wkp;  P = Pkp;  KS = 8;  }
    else             { W = Wout; P = Pout; KS = 16; }
    const int total = 16 * KS * 32 * 16;
    const int o = blockIdx.x * 256 + threadIdx.x;
    if (o >= total) return;
    const int e = o & 15, l = (o >> 4) & 31;
    const int rest = o >> 9;
    const int s = rest % KS, t = rest / KS;
    const int half = l >> 4, col = t * 16 + (l & 15);
    const int k = s * 32 + ((e < 8) ? (half * 8 + e) : (16 + half * 8 + (e - 8)));
    P[o] = (__bf16)W[k * 256 + col];
}

// ---- style -> q0 -> qc, qp, and qp . W_pos[:256] ----
__global__ __launch_bounds__(256) void k_small(
    const float* __restrict__ style, const float* __restrict__ Wd,
    const float* __restrict__ Wqc, const float* __restrict__ gqc, const float* __restrict__ bqc,
    const float* __restrict__ Wqp, const float* __restrict__ gqp, const float* __restrict__ bqp,
    const float* __restrict__ Wpos,
    float* __restrict__ qc, float* __restrict__ qp, float* __restrict__ qpd)
{
    const int n = blockIdx.x, d = threadIdx.x;
    __shared__ float q0[256];
    __shared__ float qps[256];
    float acc = 0.f;
    for (int c = 0; c < 256; ++c) acc += style[n * 256 + c] * Wd[c * 256 + d];
    acc = acc > 0.f ? acc : 0.3f * acc;          // LeakyReLU(0.3)
    q0[d] = acc;
    __syncthreads();
    float ac = 0.f, ap = 0.f;
    for (int c = 0; c < 256; ++c) { float q = q0[c]; ac += q * Wqc[c * 256 + d]; ap += q * Wqp[c * 256 + d]; }
    ac = ac * gqc[d] + bqc[d]; ac = ac > 0.f ? ac : 0.1f * ac;
    ap = ap * gqp[d] + bqp[d]; ap = ap > 0.f ? ap : 0.1f * ap;
    qc[n * 256 + d] = ac;
    qp[n * 256 + d] = ap;
    qps[d] = ap;
    __syncthreads();
    if (d < 4) {
        float s = 0.f;
        for (int c = 0; c < 256; ++c) s += qps[c] * Wpos[c * 4 + d];
        qpd[n * 4 + d] = s;
    }
}

// ---- fused 4-way CBL GEMM: Xbf[65536,256] x Wpacked, bf16 WMMA, f32 acc ----
// A tiles double-buffered via GLOBAL_LOAD_ASYNC_TO_LDS_B128; B frags straight from packed global.
// z=0: v_c -> store bf16   z=1: v_p -> store bf16
// z=2: k_c -> column sums atomically into S[n][d]
// z=3: k_p -> per-row dot with W_pos[256:,m] atomically into plog[n][m][s]
__global__ __launch_bounds__(256) void k_gemm_cbl(
    const __bf16* __restrict__ Xbf,
    const __bf16* __restrict__ Pvc, const __bf16* __restrict__ Pvp,
    const __bf16* __restrict__ Pkc, const __bf16* __restrict__ Pkp,
    const float* __restrict__ gvc, const float* __restrict__ bvc,
    const float* __restrict__ gvp, const float* __restrict__ bvp,
    const float* __restrict__ gkc, const float* __restrict__ bkc,
    const float* __restrict__ gkp, const float* __restrict__ bkp,
    const float* __restrict__ Wpos,
    __bf16* __restrict__ vc, __bf16* __restrict__ vp,
    float* __restrict__ S, float* __restrict__ plog)
{
    const int z = blockIdx.z;
    const __bf16* Wp; const float *g, *b;
    if      (z == 0) { Wp = Pvc; g = gvc; b = bvc; }
    else if (z == 1) { Wp = Pvp; g = gvp; b = bvp; }
    else if (z == 2) { Wp = Pkc; g = gkc; b = bkc; }
    else             { Wp = Pkp; g = gkp; b = bkp; }

    const int m0 = blockIdx.x * MBLK;
    const int n0 = blockIdx.y * NBLK;
    const int tid = threadIdx.x, lane = tid & 31, wave = tid >> 5;
    const int mtile = wave & 3, ntb = (wave >> 2) * 2;
    const int nt0 = (n0 >> 4) + ntb;          // global 16-col tile index

    __shared__ __align__(16) __bf16 As[2][MBLK][LDA];

    const int lr = tid >> 2;                  // 0..63 : A row
    const int lk = (tid & 3) * 8;             // 0,8,16,24
    const unsigned lds_thr = (unsigned)(unsigned long long)(const void*)&As[0][0][0]
                           + (unsigned)(lr * LDA + lk) * 2;
    const __bf16* gsrc = Xbf + (size_t)(m0 + lr) * 256 + lk;

    v8f acc0 = {}; v8f acc1 = {};

    // prologue: async-fetch tile 0 into buffer 0
    {
        unsigned long long ga = (unsigned long long)(const void*)gsrc;
        asm volatile("global_load_async_to_lds_b128 %0, %1, off" :: "v"(lds_thr), "v"(ga) : "memory");
    }
    for (int s = 0; s < 8; ++s) {
        __syncthreads();   // all waves done reading buffer (s+1)&1 (tile s-1)
        if (s + 1 < 8) {   // prefetch next tile into the other buffer
            unsigned long long ga = (unsigned long long)(const void*)(gsrc + (size_t)(s + 1) * 32);
            unsigned ld = lds_thr + (unsigned)(((s + 1) & 1) * (MBLK * LDA * 2));
            asm volatile("global_load_async_to_lds_b128 %0, %1, off" :: "v"(ld), "v"(ga) : "memory");
            asm volatile("s_wait_asynccnt 1" ::: "memory");   // tile s done (in-order), s+1 in flight
        } else {
            asm volatile("s_wait_asynccnt 0" ::: "memory");
        }
        __syncthreads();   // tile s visible to all waves

        const int half = lane >> 4, l15 = lane & 15;
        v16bf a  = load_frag(&As[s & 1][mtile * 16 + l15][0], half);
        v16bf b0 = *(const v16bf*)(Wp + ((size_t)(nt0 * 8 + s) * 32 + lane) * 16);
        v16bf b1 = *(const v16bf*)(Wp + ((size_t)((nt0 + 1) * 8 + s) * 32 + lane) * 16);
        acc0 = __builtin_amdgcn_wmma_f32_16x16x32_bf16(false, a, false, b0, (short)0, acc0, false, false);
        acc1 = __builtin_amdgcn_wmma_f32_16x16x32_bf16(false, a, false, b1, (short)0, acc1, false, false);
    }

    const int half = lane >> 4, cl = lane & 15;
    const int nb = m0 >> 12;                       // batch index (4096 rows / image)
    #pragma unroll
    for (int t = 0; t < 2; ++t) {
        const int gc = n0 + (ntb + t) * 16 + cl;
        const float gg = g[gc], bb = b[gc];
        v8f acc = t ? acc1 : acc0;
        if (z <= 1) {
            __bf16* dst = (z == 0) ? vc : vp;
            #pragma unroll
            for (int j = 0; j < 8; ++j) {
                const int row = m0 + mtile * 16 + half * 8 + j;
                float y = acc[j] * gg + bb; y = y > 0.f ? y : 0.1f * y;
                dst[(size_t)row * 256 + gc] = (__bf16)y;
            }
        } else if (z == 2) {
            float cs = 0.f;
            #pragma unroll
            for (int j = 0; j < 8; ++j) {
                float y = acc[j] * gg + bb; y = y > 0.f ? y : 0.1f * y;
                cs += y;
            }
            atomicAdd(&S[nb * 256 + gc], cs);
        } else {
            const float w0 = Wpos[(256 + gc) * 4 + 0], w1 = Wpos[(256 + gc) * 4 + 1];
            const float w2 = Wpos[(256 + gc) * 4 + 2], w3 = Wpos[(256 + gc) * 4 + 3];
            #pragma unroll
            for (int j = 0; j < 8; ++j) {
                float y = acc[j] * gg + bb; y = y > 0.f ? y : 0.1f * y;
                float p0 = y * w0, p1 = y * w1, p2 = y * w2, p3 = y * w3;
                #pragma unroll
                for (int off = 1; off < 16; off <<= 1) {   // reduce across 16 cols in each half
                    p0 += __shfl_xor(p0, off, 32);
                    p1 += __shfl_xor(p1, off, 32);
                    p2 += __shfl_xor(p2, off, 32);
                    p3 += __shfl_xor(p3, off, 32);
                }
                if (cl == 0) {
                    const int s = (m0 & 4095) + mtile * 16 + half * 8 + j;
                    atomicAdd(&plog[(nb * 4 + 0) * 4096 + s], p0);
                    atomicAdd(&plog[(nb * 4 + 1) * 4096 + s], p1);
                    atomicAdd(&plog[(nb * 4 + 2) * 4096 + s], p2);
                    atomicAdd(&plog[(nb * 4 + 3) * 4096 + s], p3);
                }
            }
        }
    }
}

// ---- channel attention: a_c = softmax_dph( scale * [4096*qc | S] @ W_ch ) ----
__global__ __launch_bounds__(256) void k_chan(
    const float* __restrict__ qc, const float* __restrict__ S,
    const float* __restrict__ Wch, float* __restrict__ a_c)
{
    const int n = blockIdx.x, c = threadIdx.x;
    float acc = 0.f;
    for (int d = 0; d < 256; ++d) acc += 4096.0f * qc[n * 256 + d] * Wch[d * 256 + c];
    for (int d = 0; d < 256; ++d) acc += S[n * 256 + d] * Wch[(256 + d) * 256 + c];
    acc *= SCALE;
    __shared__ float lg[256];
    __shared__ float mx[4], sm[4];
    lg[c] = acc;
    __syncthreads();
    if (c < 4) {                       // softmax over dph (stride-4 groups of 64)
        float m = -3.4e38f;
        for (int i = 0; i < 64; ++i) m = fmaxf(m, lg[i * 4 + c]);
        float s = 0.f;
        for (int i = 0; i < 64; ++i) s += __expf(lg[i * 4 + c] - m);
        mx[c] = m; sm[c] = s;
    }
    __syncthreads();
    a_c[n * 256 + c] = __expf(acc - mx[c & 3]) / sm[c & 3];
}

// ---- position attention: softmax over 4096 pixels per (n, head) ----
__global__ __launch_bounds__(256) void k_pos(
    const float* __restrict__ plog, const float* __restrict__ qpd,
    float* __restrict__ apos)
{
    const int n = blockIdx.x >> 2, m = blockIdx.x & 3, t = threadIdx.x;
    const float* L = plog + (n * 4 + m) * 4096;
    const float base = qpd[n * 4 + m];
    float v[16];
    float mx = -3.4e38f;
    #pragma unroll
    for (int i = 0; i < 16; ++i) { v[i] = SCALE * (base + L[i * 256 + t]); mx = fmaxf(mx, v[i]); }
    __shared__ float red[256];
    red[t] = mx; __syncthreads();
    for (int off = 128; off > 0; off >>= 1) { if (t < off) red[t] = fmaxf(red[t], red[t + off]); __syncthreads(); }
    mx = red[0]; __syncthreads();
    float s = 0.f;
    #pragma unroll
    for (int i = 0; i < 16; ++i) { v[i] = __expf(v[i] - mx); s += v[i]; }
    red[t] = s; __syncthreads();
    for (int off = 128; off > 0; off >>= 1) { if (t < off) red[t] += red[t + off]; __syncthreads(); }
    const float inv = 1.0f / red[0];
    #pragma unroll
    for (int i = 0; i < 16; ++i) {
        const int sp = i * 256 + t;
        apos[(n * 4096 + sp) * 4 + m] = v[i] * inv;
    }
}

// ---- final fused GEMM: [a_c*vc | apos*vp] (bf16) x packed W_out[512,256], CBL epilogue ----
__global__ __launch_bounds__(256) void k_final(
    const __bf16* __restrict__ vc, const __bf16* __restrict__ vp,
    const float* __restrict__ a_c, const float* __restrict__ apos,
    const __bf16* __restrict__ Pout,
    const float* __restrict__ gout, const float* __restrict__ bout,
    float* __restrict__ out)
{
    const int m0 = blockIdx.x * MBLK;
    const int n0 = blockIdx.y * NBLK;
    const int tid = threadIdx.x, lane = tid & 31, wave = tid >> 5;
    const int mtile = wave & 3, ntb = (wave >> 2) * 2;
    const int nt0 = (n0 >> 4) + ntb;
    const int nb = m0 >> 12;

    __shared__ __align__(16) __bf16 As[MBLK][LDA];

    v8f acc0 = {}; v8f acc1 = {};
    const int lr = tid >> 2, lk = (tid & 3) * 8;
    const int grow = m0 + lr;
    const int srow = grow & 4095;

    for (int s = 0; s < 16; ++s) {
        __syncthreads();
        {
            const int kg = s * 32 + lk;
            v8bf o;
            if (kg < 256) {
                v8bf r = *(const v8bf*)(vc + (size_t)grow * 256 + kg);
                const float* mul = a_c + nb * 256 + kg;
                #pragma unroll
                for (int i = 0; i < 8; ++i) o[i] = (__bf16)((float)r[i] * mul[i]);
            } else {
                const int kd = kg - 256;
                v8bf r = *(const v8bf*)(vp + (size_t)grow * 256 + kd);
                const float mul = apos[(nb * 4096 + srow) * 4 + (kd >> 6)];
                #pragma unroll
                for (int i = 0; i < 8; ++i) o[i] = (__bf16)((float)r[i] * mul);
            }
            *(v8bf*)&As[lr][lk] = o;
        }
        __syncthreads();
        const int half = lane >> 4, l15 = lane & 15;
        v16bf a  = load_frag(&As[mtile * 16 + l15][0], half);
        v16bf b0 = *(const v16bf*)(Pout + ((size_t)(nt0 * 16 + s) * 32 + lane) * 16);
        v16bf b1 = *(const v16bf*)(Pout + ((size_t)((nt0 + 1) * 16 + s) * 32 + lane) * 16);
        acc0 = __builtin_amdgcn_wmma_f32_16x16x32_bf16(false, a, false, b0, (short)0, acc0, false, false);
        acc1 = __builtin_amdgcn_wmma_f32_16x16x32_bf16(false, a, false, b1, (short)0, acc1, false, false);
    }

    const int half = lane >> 4, cl = lane & 15;
    #pragma unroll
    for (int t = 0; t < 2; ++t) {
        const int gc = n0 + (ntb + t) * 16 + cl;
        const float gg = gout[gc], bb = bout[gc];
        v8f acc = t ? acc1 : acc0;
        #pragma unroll
        for (int j = 0; j < 8; ++j) {
            const int row = m0 + mtile * 16 + half * 8 + j;
            float y = acc[j] * gg + bb; y = y > 0.f ? y : 0.1f * y;
            out[(size_t)row * 256 + gc] = y;
        }
    }
}

extern "C" void kernel_launch(void* const* d_in, const int* in_sizes, int n_in,
                              void* d_out, int out_size, void* d_ws, size_t ws_size,
                              hipStream_t stream) {
    const float* inputs  = (const float*)d_in[0];
    const float* style   = (const float*)d_in[1];
    const float* W_dense = (const float*)d_in[2];
    const float* Wq_c = (const float*)d_in[3],  *gq_c = (const float*)d_in[4],  *bq_c = (const float*)d_in[5];
    const float* Wk_c = (const float*)d_in[6],  *gk_c = (const float*)d_in[7],  *bk_c = (const float*)d_in[8];
    const float* Wv_c = (const float*)d_in[9],  *gv_c = (const float*)d_in[10], *bv_c = (const float*)d_in[11];
    const float* Wq_p = (const float*)d_in[12], *gq_p = (const float*)d_in[13], *bq_p = (const float*)d_in[14];
    const float* Wk_p = (const float*)d_in[15], *gk_p = (const float*)d_in[16], *bk_p = (const float*)d_in[17];
    const float* Wv_p = (const float*)d_in[18], *gv_p = (const float*)d_in[19], *bv_p = (const float*)d_in[20];
    const float* W_ch  = (const float*)d_in[21];
    const float* W_pos = (const float*)d_in[22];
    const float* W_out = (const float*)d_in[23];
    const float* g_out = (const float*)d_in[24];
    const float* b_out = (const float*)d_in[25];
    float* out = (float*)d_out;

    char* ws = (char*)d_ws;
    const size_t M = 65536;
    size_t off = 0;
    __bf16* vc   = (__bf16*)(ws + off); off += M * 256 * 2;          // 32 MB
    __bf16* vp   = (__bf16*)(ws + off); off += M * 256 * 2;          // 32 MB
    __bf16* Xbf  = (__bf16*)(ws + off); off += M * 256 * 2;          // 32 MB
    __bf16* Pvc  = (__bf16*)(ws + off); off += 256 * 256 * 2;        // 128 KB each
    __bf16* Pvp  = (__bf16*)(ws + off); off += 256 * 256 * 2;
    __bf16* Pkc  = (__bf16*)(ws + off); off += 256 * 256 * 2;
    __bf16* Pkp  = (__bf16*)(ws + off); off += 256 * 256 * 2;
    __bf16* Pout = (__bf16*)(ws + off); off += 512 * 256 * 2;        // 256 KB
    float*  S    = (float*)(ws + off);  off += 16 * 256 * 4;
    float*  a_c  = (float*)(ws + off);  off += 16 * 256 * 4;
    float*  plog = (float*)(ws + off);  off += 16 * 4 * 4096 * 4;    // [n][m][s]
    float*  qpd  = (float*)(ws + off);  off += 16 * 4 * 4;
    float*  apos = (float*)(ws + off);  off += 16 * 4096 * 4 * 4;    // [n][s][m]
    float*  qc   = (float*)(ws + off);  off += 16 * 256 * 4;
    float*  qp   = (float*)(ws + off);  off += 16 * 256 * 4;

    // zero the atomic accumulation buffers
    k_zero<<<dim3((16 * 256 + 255) / 256), 256, 0, stream>>>(S, 16 * 256);
    k_zero<<<dim3((16 * 4 * 4096 + 255) / 256), 256, 0, stream>>>(plog, 16 * 4 * 4096);

    // one-time data-layout preps
    k_prep_x<<<dim3(8192), 256, 0, stream>>>(inputs, Xbf);
    k_prep_w<<<dim3(512, 1, 5), 256, 0, stream>>>(Wv_c, Wv_p, Wk_c, Wk_p, W_out,
                                                  Pvc, Pvp, Pkc, Pkp, Pout);

    k_small<<<dim3(16), 256, 0, stream>>>(style, W_dense, Wq_c, gq_c, bq_c,
                                          Wq_p, gq_p, bq_p, W_pos, qc, qp, qpd);

    k_gemm_cbl<<<dim3(1024, 4, 4), 256, 0, stream>>>(
        Xbf, Pvc, Pvp, Pkc, Pkp,
        gv_c, bv_c, gv_p, bv_p, gk_c, bk_c, gk_p, bk_p,
        W_pos, vc, vp, S, plog);

    k_chan<<<dim3(16), 256, 0, stream>>>(qc, S, W_ch, a_c);
    k_pos<<<dim3(64), 256, 0, stream>>>(plog, qpd, apos);

    k_final<<<dim3(1024, 4, 1), 256, 0, stream>>>(
        vc, vp, a_c, apos, Pout, g_out, b_out, out);
}
